// DynamicFusionModule_26061861552613
// MI455X (gfx1250) — matrix operand order repaired and checked by reading, not compiled, not run on hardware
//
#include <hip/hip_runtime.h>
#include <hip/hip_bf16.h>

// ---------------------------------------------------------------------------
// CDNA5 async global->LDS path (ASYNCcnt) when the toolchain declares it
// ---------------------------------------------------------------------------
#if __has_builtin(__builtin_amdgcn_global_load_async_to_lds_b128) && \
    __has_builtin(__builtin_amdgcn_s_wait_asynccnt)
#define HAVE_ASYNC_LDS 1
#else
#define HAVE_ASYNC_LDS 0
#endif

typedef int v4i_vs __attribute__((vector_size(16)));
typedef __attribute__((address_space(1))) v4i_vs* gptr128_t;  // global v4i*
typedef __attribute__((address_space(3))) v4i_vs* lptr128_t;  // LDS v4i*

__device__ __forceinline__ void async_copy_b128(const void* g, void* l) {
#if HAVE_ASYNC_LDS
  __builtin_amdgcn_global_load_async_to_lds_b128(
      (gptr128_t)(void*)g, (lptr128_t)l, 0, 0);
#else
  (void)g; (void)l;
#endif
}

__device__ __forceinline__ void async_wait_all() {
#if HAVE_ASYNC_LDS
  __builtin_amdgcn_s_wait_asynccnt(0);
#endif
}

// ---------------------------------------------------------------------------
// Types / helpers
// ---------------------------------------------------------------------------
typedef __attribute__((ext_vector_type(16))) __bf16 v16bf;
typedef __attribute__((ext_vector_type(8)))  float  v8f;

union FragB16 {
  v16bf v;
  unsigned int u[8];
  unsigned short s[16];
};

__device__ __forceinline__ unsigned short f2bf(float f) {
  union { float f; unsigned int u; } c;
  c.f = f;
  unsigned int lsb = (c.u >> 16) & 1u;
  c.u += 0x7fffu + lsb;               // round-to-nearest-even
  return (unsigned short)(c.u >> 16);
}

static constexpr int kB    = 8;
static constexpr int kC    = 384;
static constexpr int kS    = 1024;
static constexpr long long kR = 8192;   // B*S rows
static constexpr int kHID  = 512;
static constexpr int kHD   = 96;
static constexpr int kFFN  = 1536;
static constexpr int k3C   = 1152;

// ---------------------------------------------------------------------------
// 1) transpose [B,C,S] (ir|vis) -> xt f32 [B,S,2C] and xtb bf16 [B,S,2C]
// ---------------------------------------------------------------------------
__global__ __launch_bounds__(256) void k_transpose_in(const float* __restrict__ f_ir,
                                                      const float* __restrict__ f_vis,
                                                      float* __restrict__ xt,
                                                      unsigned short* __restrict__ xtb) {
  __shared__ float tile[32][33];
  const int s0 = blockIdx.x * 32;
  const int e  = blockIdx.y / 12;
  const int c0 = (blockIdx.y % 12) * 32;
  const int b  = blockIdx.z;
  const int tx = threadIdx.x, ty = threadIdx.y;
  const float* src = (e ? f_vis : f_ir) + ((long long)b * kC + c0) * kS + s0;
#pragma unroll
  for (int j = 0; j < 4; ++j)
    tile[ty + j * 8][tx] = src[(long long)(ty + j * 8) * kS + tx];
  __syncthreads();
  const int cdst = e * kC + c0;
#pragma unroll
  for (int j = 0; j < 4; ++j) {
    int sl = ty + j * 8;
    float v = tile[tx][sl];
    long long o = ((long long)b * kS + s0 + sl) * (2 * kC) + cdst + tx;
    xt[o] = v;
    xtb[o] = f2bf(v);
  }
}

// ---------------------------------------------------------------------------
// 2) generic bf16 WMMA GEMM: C[M,N] = A[M,K] (bf16) * W[N,K]^T (f32->bf16)
//    BM=BN=64, BK=32, 128 threads (4 waves), wave w owns rows w*16..w*16+15.
// ---------------------------------------------------------------------------
struct GemmArgs {
  const unsigned short* A; long long aLd, aZ;
  const float* W;          long long wZ;      // row-major [N][K]
  const float* bias;       long long biasZ;
  void* out;               long long outLd, outZ;
  const float* res;        long long resLd, resZ;   // MODE_RES
  const float* w2;                                  // MODE_AGENT
  float* logits;                                    // MODE_AGENT
  int K;
};

constexpr int MODE_BF16  = 0;
constexpr int MODE_GELU  = 1;
constexpr int MODE_RES   = 2;
constexpr int MODE_AGENT = 3;

template <int MODE>
__global__ __launch_bounds__(128) void gemm_bf16_wmma(GemmArgs g) {
  __shared__ __align__(16) unsigned short As[64][40];   // 80B row stride
  __shared__ __align__(16) unsigned short Bs[32][72];   // 144B row stride
  const int t = threadIdx.x;
  const int w = t >> 5, l = t & 31;
  const int nlane = l & 15, half = l >> 4;
  const long long m0 = (long long)blockIdx.x * 64;
  const long long n0 = (long long)blockIdx.y * 64;
  const long long z  = blockIdx.z;
  const unsigned short* A = g.A + z * g.aZ;
  const float* W = g.W + z * g.wZ;

  v8f acc[4];
#pragma unroll
  for (int i = 0; i < 4; ++i)
#pragma unroll
    for (int r = 0; r < 8; ++r) acc[i][r] = 0.f;

  for (int k0 = 0; k0 < g.K; k0 += 32) {
    __syncthreads();
    // A tile: 64x32 bf16; async DMA to LDS when available (16B chunks)
#if HAVE_ASYNC_LDS
#pragma unroll
    for (int i = 0; i < 2; ++i) {
      int idx = t + i * 128;
      int r = idx >> 2, c8 = (idx & 3) * 8;
      async_copy_b128(A + (m0 + r) * g.aLd + k0 + c8, &As[r][c8]);
    }
#else
#pragma unroll
    for (int i = 0; i < 8; ++i) {
      int idx = t + i * 128;
      int r = idx >> 4, c2 = idx & 15;
      unsigned int val = *(const unsigned int*)(A + (m0 + r) * g.aLd + k0 + c2 * 2);
      *(unsigned int*)&As[r][c2 * 2] = val;
    }
#endif
    if (k0 + 32 < g.K)  // prefetch next A tile -> global_prefetch_b8
      __builtin_prefetch(A + (m0 + (t >> 4)) * g.aLd + k0 + 32, 0, 1);
    // B tile: transpose W[N][K] -> Bs[k][n], f32 -> bf16
#pragma unroll
    for (int i = 0; i < 16; ++i) {
      int idx = t + i * 128;
      int k = idx & 31, n = idx >> 5;
      Bs[k][n] = f2bf(W[(n0 + n) * (long long)g.K + k0 + k]);
    }
    async_wait_all();
    __syncthreads();

    FragB16 a;
    const int row = (w << 4) + nlane;
    const int kb = half << 3;
#pragma unroll
    for (int i = 0; i < 4; ++i) {
      a.u[i]     = *(const unsigned int*)&As[row][kb + 2 * i];
      a.u[4 + i] = *(const unsigned int*)&As[row][kb + 16 + 2 * i];
    }
#pragma unroll
    for (int gi = 0; gi < 4; ++gi) {
      FragB16 bb;
#pragma unroll
      for (int i = 0; i < 8; ++i) {
        bb.s[i]     = Bs[kb + i][gi * 16 + nlane];
        bb.s[8 + i] = Bs[kb + 16 + i][gi * 16 + nlane];
      }
      acc[gi] = __builtin_amdgcn_wmma_f32_16x16x32_bf16(false, a.v, false, bb.v,
                                                        (short)0, acc[gi], false, false);
    }
  }

  if (MODE == MODE_AGENT) {
    // logits[m] += sum_n silu(acc + b1[n]) * w2[n]
    float part[8];
#pragma unroll
    for (int r = 0; r < 8; ++r) part[r] = 0.f;
#pragma unroll
    for (int gi = 0; gi < 4; ++gi) {
      long long gn = n0 + gi * 16 + nlane;
      float b1 = g.bias[gn];
      float wv = g.w2[gn];
#pragma unroll
      for (int r = 0; r < 8; ++r) {
        float hv = acc[gi][r] + b1;
        float sv = hv / (1.f + __expf(-hv));
        part[r] += sv * wv;
      }
    }
#pragma unroll
    for (int r = 0; r < 8; ++r) {
      float p = part[r];
      p += __shfl_xor(p, 1, 32);
      p += __shfl_xor(p, 2, 32);
      p += __shfl_xor(p, 4, 32);
      p += __shfl_xor(p, 8, 32);
      if (nlane == 0)
        atomicAdd(g.logits + m0 + (w << 4) + r + (half << 3), p);
    }
  } else {
#pragma unroll
    for (int gi = 0; gi < 4; ++gi) {
      long long gn = n0 + gi * 16 + nlane;
      float bv = g.bias[z * g.biasZ + gn];
#pragma unroll
      for (int r = 0; r < 8; ++r) {
        long long gm = m0 + (w << 4) + r + (half << 3);
        float v = acc[gi][r] + bv;
        if (MODE == MODE_GELU) {
          v = 0.5f * v * (1.f + erff(v * 0.70710678118f));
          ((unsigned short*)g.out)[z * g.outZ + gm * g.outLd + gn] = f2bf(v);
        } else if (MODE == MODE_BF16) {
          ((unsigned short*)g.out)[z * g.outZ + gm * g.outLd + gn] = f2bf(v);
        } else {  // MODE_RES : f32 out = acc + bias + residual  (in-place safe)
          v += g.res[z * g.resZ + gm * g.resLd + gn];
          ((float*)g.out)[z * g.outZ + gm * g.outLd + gn] = v;
        }
      }
    }
  }
}

// ---------------------------------------------------------------------------
// 3) sel threshold
// ---------------------------------------------------------------------------
__global__ __launch_bounds__(256) void k_sel(const float* __restrict__ logits,
                                             const float* __restrict__ b2,
                                             float* __restrict__ sel) {
  int i = blockIdx.x * 256 + threadIdx.x;
  sel[i] = (logits[i] + b2[0] > 0.f) ? 1.f : 0.f;
}

// ---------------------------------------------------------------------------
// 4) LayerNorm over 384, wave-per-row, emits bf16
//    addr = src + (row%R)*rowStride + (row/R)*eStride
// ---------------------------------------------------------------------------
__global__ __launch_bounds__(256) void k_layernorm(const float* __restrict__ src,
                                                   long long rowStride, long long eStride,
                                                   const float* __restrict__ gamma,
                                                   const float* __restrict__ beta,
                                                   unsigned short* __restrict__ out) {
  const int rowId = blockIdx.x * 8 + (threadIdx.x >> 5);
  const int l = threadIdx.x & 31;
  const int e = (int)(rowId >= kR);
  const float* x = src + (long long)(rowId - e * kR) * rowStride + (long long)e * eStride;
  const float* gg = gamma + e * kC;
  const float* bb = beta + e * kC;
  float v[12], s = 0.f, s2 = 0.f;
#pragma unroll
  for (int i = 0; i < 12; ++i) {
    v[i] = x[l + i * 32];
    s += v[i];
    s2 += v[i] * v[i];
  }
#pragma unroll
  for (int off = 1; off < 32; off <<= 1) {
    s  += __shfl_xor(s, off, 32);
    s2 += __shfl_xor(s2, off, 32);
  }
  float mean = s * (1.f / kC);
  float var = s2 * (1.f / kC) - mean * mean;
  float inv = rsqrtf(var + 1e-5f);
#pragma unroll
  for (int i = 0; i < 12; ++i) {
    int c = l + i * 32;
    out[(long long)rowId * kC + c] = f2bf((v[i] - mean) * inv * gg[c] + bb[c]);
  }
}

// ---------------------------------------------------------------------------
// 5) flash attention: one block = 64 q rows of one (stream,batch,head)
//    grid (S/64=16, HEADS=4, 2*B=16), 128 threads (4 waves)
// ---------------------------------------------------------------------------
__global__ __launch_bounds__(128) void k_flash_attn(const unsigned short* __restrict__ qkv,
                                                    const float* __restrict__ sel,
                                                    unsigned short* __restrict__ o_bf) {
  __shared__ __align__(16) unsigned short Qs[64][104];
  __shared__ __align__(16) unsigned short KsT[96][72];
  __shared__ __align__(16) unsigned short Vs[64][104];
  __shared__ __align__(16) unsigned short Ps[64][72];
  __shared__ float sels[64];
  const int t = threadIdx.x;
  const int w = t >> 5, l = t & 31;
  const int nlane = l & 15, half = l >> 4;
  const int q0 = blockIdx.x * 64;
  const int h = blockIdx.y;
  const int eb = blockIdx.z;            // e*8 + b
  const int b = eb & 7;
  const long long rowBase = (long long)eb * kS;   // global row of s=0 (== e*R + b*S)
  const unsigned short* qPtr = qkv + rowBase * k3C + h * kHD;
  const unsigned short* kPtr = qkv + rowBase * k3C + kC + h * kHD;
  const unsigned short* vPtr = qkv + rowBase * k3C + 2 * kC + h * kHD;

  // Q tile resident: 64x96 bf16
#if HAVE_ASYNC_LDS
#pragma unroll
  for (int i = 0; i < 6; ++i) {
    int idx = t + i * 128;
    int r = idx / 12, c8 = (idx % 12) * 8;
    async_copy_b128(qPtr + (long long)(q0 + r) * k3C + c8, &Qs[r][c8]);
  }
#else
#pragma unroll
  for (int i = 0; i < 24; ++i) {
    int idx = t + i * 128;
    int r = idx / 48, c2 = idx % 48;
    *(unsigned int*)&Qs[r][c2 * 2] =
        *(const unsigned int*)(qPtr + (long long)(q0 + r) * k3C + c2 * 2);
  }
#endif

  v8f o[6];
#pragma unroll
  for (int gi = 0; gi < 6; ++gi)
#pragma unroll
    for (int r = 0; r < 8; ++r) o[gi][r] = 0.f;
  float rowmax[8], rowsum[8];
#pragma unroll
  for (int r = 0; r < 8; ++r) { rowmax[r] = -3.0e38f; rowsum[r] = 0.f; }

  const float scale = 0.1020620726f;    // 1/sqrt(96)
  const int rowq = (w << 4) + nlane;

  for (int kt = 0; kt < 16; ++kt) {
    const int k0 = kt * 64;
    __syncthreads();
    // K tile transposed into KsT[d][key] (sync; converts layout in flight)
#pragma unroll
    for (int i = 0; i < 24; ++i) {
      int idx = t + i * 128;
      int r = idx / 48, c2 = idx % 48;
      unsigned int kv = *(const unsigned int*)(kPtr + (long long)(k0 + r) * k3C + c2 * 2);
      KsT[c2 * 2][r]     = (unsigned short)(kv & 0xffffu);
      KsT[c2 * 2 + 1][r] = (unsigned short)(kv >> 16);
    }
    // V tile direct Vs[key][d]: async DMA when available
#if HAVE_ASYNC_LDS
#pragma unroll
    for (int i = 0; i < 6; ++i) {
      int idx = t + i * 128;
      int r = idx / 12, c8 = (idx % 12) * 8;
      async_copy_b128(vPtr + (long long)(k0 + r) * k3C + c8, &Vs[r][c8]);
    }
#else
#pragma unroll
    for (int i = 0; i < 24; ++i) {
      int idx = t + i * 128;
      int r = idx / 48, c2 = idx % 48;
      *(unsigned int*)&Vs[r][c2 * 2] =
          *(const unsigned int*)(vPtr + (long long)(k0 + r) * k3C + c2 * 2);
    }
#endif
    if (t < 64) sels[t] = sel[b * kS + k0 + t];
    async_wait_all();   // covers Q (first iter) + V tiles before the barrier
    __syncthreads();

    // S = Q * K^T  (M=64, N=64, K=96)
    v8f sc[4];
#pragma unroll
    for (int gi = 0; gi < 4; ++gi)
#pragma unroll
      for (int r = 0; r < 8; ++r) sc[gi][r] = 0.f;
#pragma unroll
    for (int kd = 0; kd < 96; kd += 32) {
      FragB16 a;
      int kb = kd + (half << 3);
#pragma unroll
      for (int i = 0; i < 4; ++i) {
        a.u[i]     = *(const unsigned int*)&Qs[rowq][kb + 2 * i];
        a.u[4 + i] = *(const unsigned int*)&Qs[rowq][kb + 16 + 2 * i];
      }
#pragma unroll
      for (int gi = 0; gi < 4; ++gi) {
        FragB16 bb;
#pragma unroll
        for (int i = 0; i < 8; ++i) {
          bb.s[i]     = KsT[kb + i][gi * 16 + nlane];
          bb.s[8 + i] = KsT[kb + 16 + i][gi * 16 + nlane];
        }
        sc[gi] = __builtin_amdgcn_wmma_f32_16x16x32_bf16(false, a.v, false, bb.v,
                                                         (short)0, sc[gi], false, false);
      }
    }

    // scale + key mask + tile row-max
    float tmax[8];
#pragma unroll
    for (int r = 0; r < 8; ++r) tmax[r] = -3.0e38f;
#pragma unroll
    for (int gi = 0; gi < 4; ++gi) {
      float m = sels[gi * 16 + nlane];
#pragma unroll
      for (int r = 0; r < 8; ++r) {
        float v = sc[gi][r] * scale;
        v = (m != 0.f) ? v : -1.0e9f;
        sc[gi][r] = v;
        tmax[r] = fmaxf(tmax[r], v);
      }
    }
#pragma unroll
    for (int r = 0; r < 8; ++r) {
      float v = tmax[r];
      v = fmaxf(v, __shfl_xor(v, 1, 32));
      v = fmaxf(v, __shfl_xor(v, 2, 32));
      v = fmaxf(v, __shfl_xor(v, 4, 32));
      v = fmaxf(v, __shfl_xor(v, 8, 32));
      tmax[r] = v;
    }
    float alpha[8];
#pragma unroll
    for (int r = 0; r < 8; ++r) {
      float mnew = fmaxf(rowmax[r], tmax[r]);
      alpha[r] = __expf(rowmax[r] - mnew);
      rowmax[r] = mnew;
    }
    // P = exp(S - m), tile row-sum; P -> LDS bf16 (wave-local rows)
    float tsum[8];
#pragma unroll
    for (int r = 0; r < 8; ++r) tsum[r] = 0.f;
#pragma unroll
    for (int gi = 0; gi < 4; ++gi)
#pragma unroll
      for (int r = 0; r < 8; ++r) {
        float p = __expf(sc[gi][r] - rowmax[r]);
        tsum[r] += p;
        Ps[(w << 4) + r + (half << 3)][gi * 16 + nlane] = f2bf(p);
      }
#pragma unroll
    for (int r = 0; r < 8; ++r) {
      float v = tsum[r];
      v += __shfl_xor(v, 1, 32);
      v += __shfl_xor(v, 2, 32);
      v += __shfl_xor(v, 4, 32);
      v += __shfl_xor(v, 8, 32);
      rowsum[r] = rowsum[r] * alpha[r] + v;
    }
#pragma unroll
    for (int gi = 0; gi < 6; ++gi)
#pragma unroll
      for (int r = 0; r < 8; ++r) o[gi][r] *= alpha[r];

    // O += P * V   (M=64, N=96, K=64)
#pragma unroll
    for (int ks = 0; ks < 64; ks += 32) {
      FragB16 a;
      int kb = ks + (half << 3);
#pragma unroll
      for (int i = 0; i < 4; ++i) {
        a.u[i]     = *(const unsigned int*)&Ps[rowq][kb + 2 * i];
        a.u[4 + i] = *(const unsigned int*)&Ps[rowq][kb + 16 + 2 * i];
      }
#pragma unroll
      for (int gi = 0; gi < 6; ++gi) {
        FragB16 bb;
#pragma unroll
        for (int i = 0; i < 8; ++i) {
          bb.s[i]     = Vs[kb + i][gi * 16 + nlane];
          bb.s[8 + i] = Vs[kb + 16 + i][gi * 16 + nlane];
        }
        o[gi] = __builtin_amdgcn_wmma_f32_16x16x32_bf16(false, a.v, false, bb.v,
                                                        (short)0, o[gi], false, false);
      }
    }
  }

  // normalize and store bf16 into o_bf [2R][384]
#pragma unroll
  for (int gi = 0; gi < 6; ++gi)
#pragma unroll
    for (int r = 0; r < 8; ++r) {
      int mm = (w << 4) + r + (half << 3);
      float v = o[gi][r] / rowsum[r];
      long long orow = rowBase + q0 + mm;
      o_bf[orow * kC + h * kHD + gi * 16 + nlane] = f2bf(v);
    }
}

// ---------------------------------------------------------------------------
// 6) finalize: where(sel, y_ir+y_vis, base) transposed back to [B,C,H,W]
// ---------------------------------------------------------------------------
__global__ __launch_bounds__(256) void k_finalize(const float* __restrict__ x1,
                                                  const float* __restrict__ xt,
                                                  const float* __restrict__ sel,
                                                  float* __restrict__ out) {
  __shared__ float tile[32][33];
  const int s0 = blockIdx.x * 32;
  const int c0 = blockIdx.y * 32;
  const int b = blockIdx.z;
  const int tx = threadIdx.x, ty = threadIdx.y;
#pragma unroll
  for (int j = 0; j < 4; ++j) {
    int sl = ty + j * 8;
    long long r = (long long)b * kS + s0 + sl;
    int c = c0 + tx;
    float refined = x1[r * kC + c] + x1[kR * kC + r * kC + c];
    float base = xt[r * (2 * kC) + c] + xt[r * (2 * kC) + kC + c];
    tile[sl][tx] = (sel[r] != 0.f) ? refined : base;
  }
  __syncthreads();
#pragma unroll
  for (int j = 0; j < 4; ++j) {
    int cl = ty + j * 8;
    out[((long long)b * kC + c0 + cl) * kS + s0 + tx] = tile[tx][cl];
  }
}

// ---------------------------------------------------------------------------
// launcher
// ---------------------------------------------------------------------------
extern "C" void kernel_launch(void* const* d_in, const int* in_sizes, int n_in,
                              void* d_out, int out_size, void* d_ws, size_t ws_size,
                              hipStream_t stream) {
  (void)in_sizes; (void)n_in; (void)out_size; (void)ws_size;
  const float* f_ir     = (const float*)d_in[0];
  const float* f_vis    = (const float*)d_in[1];
  const float* agent_w1 = (const float*)d_in[2];
  const float* agent_b1 = (const float*)d_in[3];
  const float* agent_w2 = (const float*)d_in[4];
  const float* agent_b2 = (const float*)d_in[5];
  const float* norm_g   = (const float*)d_in[6];
  const float* norm_b   = (const float*)d_in[7];
  const float* in_w     = (const float*)d_in[8];
  const float* in_b     = (const float*)d_in[9];
  const float* out_w    = (const float*)d_in[10];
  const float* out_b    = (const float*)d_in[11];
  const float* ffn_w1   = (const float*)d_in[12];
  const float* ffn_b1   = (const float*)d_in[13];
  const float* ffn_w2   = (const float*)d_in[14];
  const float* ffn_b2   = (const float*)d_in[15];
  float* out = (float*)d_out;

  char* ws = (char*)d_ws;
  // workspace layout (bytes); lifetime-based pool reuse
  float* xt = (float*)ws;                                         // 25,165,824
  float* x1 = (float*)(ws + 25165824LL);                          // 25,165,824
  float* logits = (float*)(ws + 50331648LL);                      //     32,768
  float* sel    = (float*)(ws + 50364416LL);                      //     32,768
  unsigned short* poolA = (unsigned short*)(ws + 50397184LL);     // 12,582,912
  unsigned short* xtb    = poolA;          // agent A operand
  unsigned short* o_bf   = poolA;          // reuse after agent done
  unsigned short* xn2_bf = poolA;          // reuse after out-proj done
  char* poolB = ws + 62980096LL;                                  // 50,331,648
  unsigned short* xn_bf  = (unsigned short*)poolB;
  unsigned short* qkv_bf = (unsigned short*)(poolB + 12582912LL);
  unsigned short* ff1_bf = (unsigned short*)poolB;  // reuse after flash

  // 1) transpose + bf16 copy
  k_transpose_in<<<dim3(32, 24, kB), dim3(32, 8), 0, stream>>>(f_ir, f_vis, xt, xtb);

  // 2) agent GEMM + fused SiLU/w2 reduction
  (void)hipMemsetAsync(logits, 0, kR * sizeof(float), stream);
  {
    GemmArgs a{};
    a.A = xtb; a.aLd = 2 * kC; a.aZ = 0;
    a.W = agent_w1; a.wZ = 0;
    a.bias = agent_b1; a.biasZ = 0;
    a.w2 = agent_w2; a.logits = logits;
    a.K = 2 * kC;
    gemm_bf16_wmma<MODE_AGENT><<<dim3(128, kHID / 64, 1), 128, 0, stream>>>(a);
  }
  k_sel<<<dim3(kR / 256), 256, 0, stream>>>(logits, agent_b2, sel);

  // 3) LN1 -> xn_bf
  k_layernorm<<<dim3(2 * kR / 8), 256, 0, stream>>>(xt, 2 * kC, kC, norm_g, norm_b, xn_bf);

  // 4) QKV GEMM (batched over streams via grid.z)
  {
    GemmArgs a{};
    a.A = xn_bf; a.aLd = kC; a.aZ = kR * kC;
    a.W = in_w; a.wZ = (long long)k3C * kC;
    a.bias = in_b; a.biasZ = k3C;
    a.out = qkv_bf; a.outLd = k3C; a.outZ = kR * k3C;
    a.K = kC;
    gemm_bf16_wmma<MODE_BF16><<<dim3(128, k3C / 64, 2), 128, 0, stream>>>(a);
  }

  // 5) flash attention -> o_bf
  k_flash_attn<<<dim3(kS / 64, 4, 16), 128, 0, stream>>>(qkv_bf, sel, o_bf);

  // 6) out-proj + residual -> x1 (f32)
  {
    GemmArgs a{};
    a.A = o_bf; a.aLd = kC; a.aZ = kR * kC;
    a.W = out_w; a.wZ = (long long)kC * kC;
    a.bias = out_b; a.biasZ = kC;
    a.out = x1; a.outLd = kC; a.outZ = kR * kC;
    a.res = xt; a.resLd = 2 * kC; a.resZ = kC;
    a.K = kC;
    gemm_bf16_wmma<MODE_RES><<<dim3(128, kC / 64, 2), 128, 0, stream>>>(a);
  }

  // 7) LN2 -> xn2_bf
  k_layernorm<<<dim3(2 * kR / 8), 256, 0, stream>>>(x1, kC, kR * kC, norm_g, norm_b, xn2_bf);

  // 8) FFN1 + GELU -> ff1_bf
  {
    GemmArgs a{};
    a.A = xn2_bf; a.aLd = kC; a.aZ = kR * kC;
    a.W = ffn_w1; a.wZ = (long long)kFFN * kC;
    a.bias = ffn_b1; a.biasZ = kFFN;
    a.out = ff1_bf; a.outLd = kFFN; a.outZ = kR * kFFN;
    a.K = kC;
    gemm_bf16_wmma<MODE_GELU><<<dim3(128, kFFN / 64, 2), 128, 0, stream>>>(a);
  }

  // 9) FFN2 + residual, in-place y = x1 + ffn
  {
    GemmArgs a{};
    a.A = ff1_bf; a.aLd = kFFN; a.aZ = kR * kFFN;
    a.W = ffn_w2; a.wZ = (long long)kC * kFFN;
    a.bias = ffn_b2; a.biasZ = kC;
    a.out = x1; a.outLd = kC; a.outZ = kR * kC;
    a.res = x1; a.resLd = kC; a.resZ = kR * kC;
    a.K = kFFN;
    gemm_bf16_wmma<MODE_RES><<<dim3(128, kC / 64, 2), 128, 0, stream>>>(a);
  }

  // 10) select + transpose back to [B,C,H,W]
  k_finalize<<<dim3(32, 12, kB), dim3(32, 8), 0, stream>>>(x1, xt, sel, out);
}
